// EfficientAttnApprox_63513976373282
// MI455X (gfx1250) — matrix operand order repaired and checked by reading, not compiled, not run on hardware
//
#include <hip/hip_runtime.h>
#include <hip/hip_bf16.h>
#include <stdint.h>

typedef __attribute__((ext_vector_type(16))) __bf16 v16bf;
typedef __attribute__((ext_vector_type(8)))  __bf16 v8bf;
typedef __attribute__((ext_vector_type(4)))  __bf16 v4bf;
typedef __attribute__((ext_vector_type(8)))  float  v8f;
typedef __attribute__((ext_vector_type(4)))  float  v4f;
typedef __attribute__((ext_vector_type(2)))  float  v2f;

#define BB 4
#define SS 4096
#define DD 2048
#define KK 64
#define HH 32
#define NROWS (BB*SS)      // 16384
#define NQKV 192
#define CHUNK 512
#define LDA (CHUNK + 8)    // padded row stride (halves): 1040B = 65*16, conflict-free
#define LDATT 72           // padded row stride for attn tile: 144B = 9*16

// ---------------- Kernel A: weight conversion + KV-state zero ----------------
__global__ __launch_bounds__(256)
void k_prep(const float* __restrict__ wq, const float* __restrict__ wo,
            __bf16* __restrict__ wq_bf, __bf16* __restrict__ wo_bf,
            float* __restrict__ kvs)
{
    int i = blockIdx.x * 256 + threadIdx.x;
    if (i < NQKV * DD) wq_bf[i] = (__bf16)wq[i];
    if (i < DD * KK)   wo_bf[i] = (__bf16)wo[i];
    if (i < BB * HH * 6) kvs[i] = 0.f;
}

// ------------- Kernel B: LayerNorm + QKV GEMM (bf16 WMMA) -------------------
__global__ __launch_bounds__(256)
void k_ln_qkv(const float* __restrict__ x, const float* __restrict__ nw,
              const float* __restrict__ nb, const __bf16* __restrict__ wq,
              const float* __restrict__ bq, float* __restrict__ qkv)
{
    __shared__ __bf16 lA[32 * LDA];
    const int wave = threadIdx.x >> 5;
    const int lane = threadIdx.x & 31;
    const int row0 = blockIdx.x * 32;

    // pass 1: LN statistics for this wave's 4 tokens
    float mu_[4], rs_[4];
    for (int t = 0; t < 4; ++t) {
        const float* xr = x + (size_t)(row0 + wave * 4 + t) * DD;
        float s = 0.f, s2 = 0.f;
        #pragma unroll
        for (int i = 0; i < 16; ++i) {
            v4f v = *(const v4f*)(xr + i * 128 + lane * 4);
            #pragma unroll
            for (int j = 0; j < 4; ++j) { s += v[j]; s2 += v[j] * v[j]; }
        }
        #pragma unroll
        for (int off = 16; off > 0; off >>= 1) {
            s  += __shfl_xor(s,  off, 32);
            s2 += __shfl_xor(s2, off, 32);
        }
        float mu  = s * (1.f / DD);
        float var = s2 * (1.f / DD) - mu * mu;
        mu_[t] = mu;
        rs_[t] = rsqrtf(var + 1e-5f);
    }

    const int mtile = wave >> 2;    // 2 m-tiles of 16 tokens
    const int wcol  = wave & 3;     // 4 wave-columns x 3 n-tiles = 192 cols
    const int hf    = lane >> 4;
    const int r16   = lane & 15;
    v8f acc0 = {}, acc1 = {}, acc2 = {};

    for (int c = 0; c < DD; c += CHUNK) {
        // normalize this K-chunk into LDS (bf16)
        for (int t = 0; t < 4; ++t) {
            const int trow = wave * 4 + t;
            const float* xr = x + (size_t)(row0 + trow) * DD + c;
            __bf16* dst = lA + trow * LDA;
            const float mu = mu_[t], rs = rs_[t];
            #pragma unroll
            for (int i = 0; i < 4; ++i) {
                int k = i * 128 + lane * 4;
                v4f v  = *(const v4f*)(xr + k);
                v4f w  = *(const v4f*)(nw + c + k);
                v4f b2 = *(const v4f*)(nb + c + k);
                v4bf o;
                #pragma unroll
                for (int j = 0; j < 4; ++j)
                    o[j] = (__bf16)((v[j] - mu) * rs * w[j] + b2[j]);
                *(v4bf*)(dst + k) = o;
            }
        }
        __syncthreads();

        // GEMM over chunk: 16 WMMA K-steps
        for (int kb = 0; kb < CHUNK; kb += 32) {
            const __bf16* ap = lA + (mtile * 16 + r16) * LDA + kb + hf * 16;
            v8bf c0 = *(const v8bf*)ap;
            v8bf c1 = *(const v8bf*)(ap + 8);
            v16bf a;
            #pragma unroll
            for (int i = 0; i < 8; ++i) { a[i] = c0[i]; a[8 + i] = c1[i]; }
            const int kglb = c + kb + hf * 16;
            {
                v16bf bm = *(const v16bf*)(wq + (size_t)((wcol * 3 + 0) * 16 + r16) * DD + kglb);
                acc0 = __builtin_amdgcn_wmma_f32_16x16x32_bf16(false, a, false, bm, (short)0, acc0, false, false);
            }
            {
                v16bf bm = *(const v16bf*)(wq + (size_t)((wcol * 3 + 1) * 16 + r16) * DD + kglb);
                acc1 = __builtin_amdgcn_wmma_f32_16x16x32_bf16(false, a, false, bm, (short)0, acc1, false, false);
            }
            {
                v16bf bm = *(const v16bf*)(wq + (size_t)((wcol * 3 + 2) * 16 + r16) * DD + kglb);
                acc2 = __builtin_amdgcn_wmma_f32_16x16x32_bf16(false, a, false, bm, (short)0, acc2, false, false);
            }
        }
        __syncthreads();
    }

    // epilogue: + bias, store fp32 qkv scratch
    const int mbase = row0 + mtile * 16 + hf * 8;
    #pragma unroll
    for (int j = 0; j < 3; ++j) {
        v8f av = (j == 0) ? acc0 : ((j == 1) ? acc1 : acc2);
        const int n = (wcol * 3 + j) * 16 + r16;
        const float bias = bq[n];
        #pragma unroll
        for (int r = 0; r < 8; ++r)
            qkv[(size_t)(mbase + r) * NQKV + n] = av[r] + bias;
    }
}

// ------- Kernel C: feature map (ReLU 2x2) + hierarchical KV reduction -------
__global__ __launch_bounds__(256)
void k_fm_kv(const float* __restrict__ qkv, const float* __restrict__ fmw,
             const float* __restrict__ fmb, float* __restrict__ qp,
             float* __restrict__ kvs)
{
    __shared__ float lkv[HH * 6];
    if (threadIdx.x < HH * 6) lkv[threadIdx.x] = 0.f;
    __syncthreads();
    const int wave = threadIdx.x >> 5;
    const int h    = threadIdx.x & 31;          // one head per lane
    const int t    = blockIdx.x * 8 + wave;     // one token per wave
    const float* row = qkv + (size_t)t * NQKV;
    v2f q = *(const v2f*)(row + h * 2);
    v2f k = *(const v2f*)(row + 64 + h * 2);
    v2f v = *(const v2f*)(row + 128 + h * 2);
    const float w00 = fmw[0], w01 = fmw[1], w10 = fmw[2], w11 = fmw[3];
    const float b0 = fmb[0], b1 = fmb[1];
    float qp0 = fmaxf(w00 * q[0] + w01 * q[1] + b0, 0.f);
    float qp1 = fmaxf(w10 * q[0] + w11 * q[1] + b1, 0.f);
    float kp0 = fmaxf(w00 * k[0] + w01 * k[1] + b0, 0.f);
    float kp1 = fmaxf(w10 * k[0] + w11 * k[1] + b1, 0.f);
    v2f qo; qo[0] = qp0; qo[1] = qp1;
    *(v2f*)(qp + (size_t)t * KK + h * 2) = qo;
    atomicAdd(&lkv[h * 6 + 0], kp0 * v[0]);
    atomicAdd(&lkv[h * 6 + 1], kp0 * v[1]);
    atomicAdd(&lkv[h * 6 + 2], kp1 * v[0]);
    atomicAdd(&lkv[h * 6 + 3], kp1 * v[1]);
    atomicAdd(&lkv[h * 6 + 4], kp0);
    atomicAdd(&lkv[h * 6 + 5], kp1);
    __syncthreads();
    if (threadIdx.x < HH * 6) {
        int b = (blockIdx.x * 8) >> 12;   // token / 4096
        atomicAdd(&kvs[(size_t)b * HH * 6 + threadIdx.x], lkv[threadIdx.x]);
    }
}

// ---- Kernel D: attn = q'KV/(q'.ksum) -> out GEMM (bf16 WMMA) + residual ----
__global__ __launch_bounds__(256)
void k_attn_out(const float* __restrict__ qp, const float* __restrict__ kvs,
                const __bf16* __restrict__ wo, const float* __restrict__ bo,
                const float* __restrict__ x, float* __restrict__ out)
{
    __shared__ __bf16 lA[32 * LDATT];
    const int wave = threadIdx.x >> 5;
    const int lane = threadIdx.x & 31;
    const int row0 = blockIdx.x * 32;
    const int n0   = blockIdx.y * 256;
    const int b    = row0 >> 12;

    // stage attn tile (32 tokens x 64) as bf16 A-matrix
    for (int t = 0; t < 4; ++t) {
        const int trow = wave * 4 + t;
        const int tok  = row0 + trow;
        const int h    = lane;
        v2f q = *(const v2f*)(qp + (size_t)tok * KK + h * 2);
        const float* kv = kvs + (size_t)(b * HH + h) * 6;
        float denom = q[0] * kv[4] + q[1] * kv[5] + 1e-8f;
        float rd = 1.f / denom;
        float a0 = (q[0] * kv[0] + q[1] * kv[2]) * rd;
        float a1 = (q[0] * kv[1] + q[1] * kv[3]) * rd;
        lA[trow * LDATT + h * 2]     = (__bf16)a0;
        lA[trow * LDATT + h * 2 + 1] = (__bf16)a1;
    }
    __syncthreads();

    const int mtile = wave >> 2;
    const int wcol  = wave & 3;     // 4 wave-cols x 4 n-tiles = 256 cols
    const int hf    = lane >> 4;
    const int r16   = lane & 15;
    v8f acc[4] = {};
    #pragma unroll
    for (int ks = 0; ks < 2; ++ks) {
        const int kb = ks * 32;
        const __bf16* ap = lA + (mtile * 16 + r16) * LDATT + kb + hf * 16;
        v8bf c0 = *(const v8bf*)ap;
        v8bf c1 = *(const v8bf*)(ap + 8);
        v16bf a;
        #pragma unroll
        for (int i = 0; i < 8; ++i) { a[i] = c0[i]; a[8 + i] = c1[i]; }
        #pragma unroll
        for (int j = 0; j < 4; ++j) {
            const int n = n0 + (wcol * 4 + j) * 16 + r16;
            v16bf bm = *(const v16bf*)(wo + (size_t)n * KK + kb + hf * 16);
            acc[j] = __builtin_amdgcn_wmma_f32_16x16x32_bf16(false, a, false, bm, (short)0, acc[j], false, false);
        }
    }
    // epilogue: + out bias + residual x
    const int mbase = row0 + mtile * 16 + hf * 8;
    #pragma unroll
    for (int j = 0; j < 4; ++j) {
        const int n = n0 + (wcol * 4 + j) * 16 + r16;
        const float bias = bo[n];
        #pragma unroll
        for (int r = 0; r < 8; ++r) {
            size_t idx = (size_t)(mbase + r) * DD + n;
            out[idx] = acc[j][r] + bias + x[idx];
        }
    }
}

extern "C" void kernel_launch(void* const* d_in, const int* in_sizes, int n_in,
                              void* d_out, int out_size, void* d_ws, size_t ws_size,
                              hipStream_t stream)
{
    const float* x   = (const float*)d_in[0];
    const float* nw  = (const float*)d_in[1];
    const float* nb  = (const float*)d_in[2];
    const float* wq  = (const float*)d_in[3];
    const float* bq  = (const float*)d_in[4];
    const float* fmw = (const float*)d_in[5];
    const float* fmb = (const float*)d_in[6];
    const float* wo  = (const float*)d_in[7];
    const float* bo  = (const float*)d_in[8];
    float* out = (float*)d_out;

    char* ws = (char*)d_ws;
    __bf16* wq_bf = (__bf16*)(ws);                 //   786,432 B
    __bf16* wo_bf = (__bf16*)(ws + 786432);        //   262,144 B
    float*  qkv   = (float*)(ws + 1048576);        // 12,582,912 B
    float*  qpbuf = (float*)(ws + 13631488);       //  4,194,304 B
    float*  kvs   = (float*)(ws + 17825792);       //      3,072 B

    k_prep   <<<1536, 256, 0, stream>>>(wq, wo, wq_bf, wo_bf, kvs);
    k_ln_qkv <<<NROWS / 32, 256, 0, stream>>>(x, nw, nb, wq_bf, bq, qkv);
    k_fm_kv  <<<NROWS / 8, 256, 0, stream>>>(qkv, fmw, fmb, qpbuf, kvs);
    k_attn_out<<<dim3(NROWS / 32, DD / 256), 256, 0, stream>>>(qpbuf, kvs, wo_bf, bo, x, out);
}
